// ScaledDotProductAttention_76364518523288
// MI455X (gfx1250) — matrix operand order repaired and checked by reading, not compiled, not run on hardware
//
#include <hip/hip_runtime.h>

typedef __attribute__((ext_vector_type(16))) __bf16 v16bf;
typedef __attribute__((ext_vector_type(8)))  __bf16 v8bf;
typedef __attribute__((ext_vector_type(8)))  float  v8f;

#define BATCH_H 64     // B*H = 4*16
#define SEQ     2048
#define DIM     64
#define QTILE   64     // query rows per block (4 waves x 16)
#define KTILE   64     // keys per inner iteration (4 x 16-key score tiles)
#define KPITCH  72     // bf16 elems per K row  (144B pitch, 16B-aligned)
#define VPITCH  72     // bf16 elems per Vt row (64 keys + pad)
#define PPITCH  72     // bf16 elems per P row  (64 keys + pad)

static __device__ inline v16bf cat8(v8bf a, v8bf b) {
    return __builtin_shufflevector(a, b, 0,1,2,3,4,5,6,7,8,9,10,11,12,13,14,15);
}
// XOR-swizzle within 16-lane groups: and_mask=0x1f, or=0, xor=d
#define SWIZ_XOR(x, d) \
    __int_as_float(__builtin_amdgcn_ds_swizzle(__float_as_int(x), 0x1f | ((d) << 10)))

__global__ __launch_bounds__(128)
void fa_causal_wmma_kernel(const float* __restrict__ Q,
                           const float* __restrict__ K,
                           const float* __restrict__ V,
                           float* __restrict__ O)
{
    const int tid  = threadIdx.x;
    const int lane = tid & 31;
    const int wave = tid >> 5;
    const int lh   = lane >> 4;        // lane half
    const int ln   = lane & 15;        // lane-in-half (N / M index)

    const int bh    = blockIdx.y;
    const int qb    = blockIdx.x * QTILE;
    const int qbase = qb + wave * 16;  // first query row of this wave

    const size_t bhoff = (size_t)bh * SEQ * DIM;
    const float* qp = Q + bhoff;
    const float* kp = K + bhoff;
    const float* vp = V + bhoff;
    float*       op = O + bhoff;

    __shared__ __align__(16) __bf16 kbuf[2][KTILE][KPITCH];  // K tile, row-major
    __shared__ __align__(16) __bf16 vbuf[2][DIM][VPITCH];    // V tile, TRANSPOSED [d][key]
    __shared__ __align__(16) __bf16 pbuf[4][16][PPITCH];     // per-wave P scratch (16x64)

    // ---- Q as two 16x32 bf16 A-fragments, 1/sqrt(D) folded in ----
    const float scale = 0.125f;
    const int qrowA = qbase + ln;
    v16bf qa[2];
#pragma unroll
    for (int h = 0; h < 2; ++h) {
        const float4* r0 = (const float4*)&qp[(size_t)qrowA * DIM + 32*h + 16*lh];
        const float4* r1 = (const float4*)&qp[(size_t)qrowA * DIM + 32*h + 16*lh + 8];
        float4 a0 = r0[0], a1 = r0[1];
        float4 b0 = r1[0], b1 = r1[1];
        qa[h][0]=(__bf16)(a0.x*scale);  qa[h][1]=(__bf16)(a0.y*scale);
        qa[h][2]=(__bf16)(a0.z*scale);  qa[h][3]=(__bf16)(a0.w*scale);
        qa[h][4]=(__bf16)(a1.x*scale);  qa[h][5]=(__bf16)(a1.y*scale);
        qa[h][6]=(__bf16)(a1.z*scale);  qa[h][7]=(__bf16)(a1.w*scale);
        qa[h][8]=(__bf16)(b0.x*scale);  qa[h][9]=(__bf16)(b0.y*scale);
        qa[h][10]=(__bf16)(b0.z*scale); qa[h][11]=(__bf16)(b0.w*scale);
        qa[h][12]=(__bf16)(b1.x*scale); qa[h][13]=(__bf16)(b1.y*scale);
        qa[h][14]=(__bf16)(b1.z*scale); qa[h][15]=(__bf16)(b1.w*scale);
    }

    // all-ones B fragment: row-sum chunks (l accumulated on the matrix pipe)
    v16bf ones;
#pragma unroll
    for (int e = 0; e < 16; ++e) ones[e] = (__bf16)1.0f;

    // ---- state ----
    v8f acc[4], accl, vzero;
#pragma unroll
    for (int i = 0; i < 8; ++i) vzero[i] = 0.0f;
#pragma unroll
    for (int c = 0; c < 4; ++c) acc[c] = vzero;
    accl = vzero;
    float mrun[8];
#pragma unroll
    for (int i = 0; i < 8; ++i) mrun[i] = -3.0e30f;

    // ---- cooperative staging geometry (affine: one base + imm offsets) ----
    const int skey = tid >> 5;          // K: base key row (advances +4 per step)
    const int sdp  = tid & 31;          // K: d-pair
    const int svd  = tid & 63;          // V: d (constant across steps)
    const int svk  = (tid >> 6) * 2;    // V: base key (advances +4 per step)

    auto stage = [&](int buf, int jb) {
        const float* kst = kp + (size_t)(jb + skey) * DIM + 2 * sdp;
        const float* vst = vp + (size_t)(jb + svk)  * DIM + svd;
        __bf16* kdst = &kbuf[buf][skey][2 * sdp];
        __bf16* vdst = &vbuf[buf][svd][svk];
#pragma unroll
        for (int i = 0; i < 16; ++i) {          // K: 64x64, 4 rows per step
            float2 f = *(const float2*)(kst + (size_t)i * 4 * DIM);
            kdst[i * 4 * KPITCH]     = (__bf16)f.x;
            kdst[i * 4 * KPITCH + 1] = (__bf16)f.y;
        }
#pragma unroll
        for (int i = 0; i < 16; ++i) {          // Vt[d][key], 4 keys per step
            float a = vst[(size_t)i * 4 * DIM];
            float b = vst[(size_t)i * 4 * DIM + DIM];
            vdst[i * 4]     = (__bf16)a;
            vdst[i * 4 + 1] = (__bf16)b;
        }
        if (jb + 2 * KTILE <= SEQ) {            // hint next tile toward L2
            __builtin_prefetch(kst + (size_t)KTILE * DIM, 0, 1);
            __builtin_prefetch(vst + (size_t)KTILE * DIM, 0, 1);
        }
    };

    const int qlast = qbase + 15;               // causal bound of this wave
    const int niter = qb / KTILE + 1;           // block-wide trip count

    stage(0, 0);

    for (int it = 0; it < niter; ++it) {
        const int jb  = it * KTILE;
        __syncthreads();                        // staged tile `it` visible; prior reads done
        const int cur = it & 1;
        if (it + 1 < niter) stage(1 - cur, jb + KTILE);
        if (jb > qlast) continue;               // fully-masked for this wave (uniform)

        // ---- scores: four 16x16 tiles (keys jb+16t .. jb+16t+15) ----
        v8f ct[4];
#pragma unroll
        for (int t = 0; t < 4; ++t) {
            const __bf16* krow = &kbuf[cur][16 * t + ln][16 * lh];
            v16bf kb0 = cat8(*(const v8bf*)(krow),      *(const v8bf*)(krow + 8));
            v16bf kb1 = cat8(*(const v8bf*)(krow + 32), *(const v8bf*)(krow + 40));
            v8f cs = __builtin_amdgcn_wmma_f32_16x16x32_bf16(
                        false, qa[0], false, kb0, (short)0, vzero, false, false);
            ct[t]  = __builtin_amdgcn_wmma_f32_16x16x32_bf16(
                        false, qa[1], false, kb1, (short)0, cs, false, false);
        }

        // ---- causal mask (diagonal tile only; wave-uniform predicate) ----
        if (jb + KTILE - 1 > qbase) {
#pragma unroll
            for (int i = 0; i < 8; ++i) {
                const int qr = qbase + i + 8 * lh;
#pragma unroll
                for (int t = 0; t < 4; ++t)
                    if (jb + 16 * t + ln > qr) ct[t][i] = -3.0e30f;
            }
        }

        // ---- online softmax: ONE reduction/rescale for all 64 keys ----
        float alpha[8];
#pragma unroll
        for (int i = 0; i < 8; ++i) {
            float mx = fmaxf(fmaxf(ct[0][i], ct[1][i]), fmaxf(ct[2][i], ct[3][i]));
            mx = fmaxf(mx, SWIZ_XOR(mx, 1));
            mx = fmaxf(mx, SWIZ_XOR(mx, 2));
            mx = fmaxf(mx, SWIZ_XOR(mx, 4));
            mx = fmaxf(mx, SWIZ_XOR(mx, 8));
            const float mnew = fmaxf(mrun[i], mx);
            alpha[i] = __expf(mrun[i] - mnew);
#pragma unroll
            for (int t = 0; t < 4; ++t) ct[t][i] = __expf(ct[t][i] - mnew);
            mrun[i] = mnew;
        }
#pragma unroll
        for (int c = 0; c < 4; ++c)
#pragma unroll
            for (int i = 0; i < 8; ++i) acc[c][i] *= alpha[i];
#pragma unroll
        for (int i = 0; i < 8; ++i) accl[i] *= alpha[i];

        // ---- P (16x64): C layout -> A layout through per-wave LDS scratch ----
#pragma unroll
        for (int i = 0; i < 8; ++i) {
            const int m = i + 8 * lh;
#pragma unroll
            for (int t = 0; t < 4; ++t)
                pbuf[wave][m][16 * t + ln] = (__bf16)ct[t][i];
        }
        const __bf16* prow = &pbuf[wave][ln][16 * lh];
        v16bf pa0 = cat8(*(const v8bf*)(prow),      *(const v8bf*)(prow + 8));
        v16bf pa1 = cat8(*(const v8bf*)(prow + 32), *(const v8bf*)(prow + 40));

        // ---- O += P(16x64) x V(64x64); l += P x 1 ----
#pragma unroll
        for (int c = 0; c < 4; ++c) {
            const __bf16* vrow = &vbuf[cur][16 * c + ln][16 * lh];
            v16bf vb0 = cat8(*(const v8bf*)(vrow),      *(const v8bf*)(vrow + 8));
            v16bf vb1 = cat8(*(const v8bf*)(vrow + 32), *(const v8bf*)(vrow + 40));
            acc[c] = __builtin_amdgcn_wmma_f32_16x16x32_bf16(
                        false, pa0, false, vb0, (short)0, acc[c], false, false);
            acc[c] = __builtin_amdgcn_wmma_f32_16x16x32_bf16(
                        false, pa1, false, vb1, (short)0, acc[c], false, false);
        }
        accl = __builtin_amdgcn_wmma_f32_16x16x32_bf16(
                    false, pa0, false, ones, (short)0, accl, false, false);
        accl = __builtin_amdgcn_wmma_f32_16x16x32_bf16(
                    false, pa1, false, ones, (short)0, accl, false, false);
    }

    // ---- epilogue: normalize and store ----
#pragma unroll
    for (int i = 0; i < 8; ++i) {
        const int m = i + 8 * lh;
        const float inv = 1.0f / accl[i];
        float* orow = &op[(size_t)(qbase + m) * DIM + ln];
        orow[0]  = acc[0][i] * inv;
        orow[16] = acc[1][i] * inv;
        orow[32] = acc[2][i] * inv;
        orow[48] = acc[3][i] * inv;
    }
}

extern "C" void kernel_launch(void* const* d_in, const int* in_sizes, int n_in,
                              void* d_out, int out_size, void* d_ws, size_t ws_size,
                              hipStream_t stream) {
    (void)in_sizes; (void)n_in; (void)d_ws; (void)ws_size; (void)out_size;
    const float* q = (const float*)d_in[0];
    const float* k = (const float*)d_in[1];
    const float* v = (const float*)d_in[2];
    // d_in[3] is the causal mask; causality is computed analytically in-kernel.
    float* o = (float*)d_out;

    dim3 grid(SEQ / QTILE, BATCH_H);
    dim3 block(128);
    hipLaunchKernelGGL(fa_causal_wmma_kernel, grid, block, 0, stream, q, k, v, o);
}
